// MultiHeadAttention_65283502899610
// MI455X (gfx1250) — matrix operand order repaired and checked by reading, not compiled, not run on hardware
//
#include <hip/hip_runtime.h>
#include <hip/hip_bf16.h>

typedef __bf16 bf16_t;
typedef __attribute__((ext_vector_type(16))) __bf16 v16bf;
typedef __attribute__((ext_vector_type(8)))  __bf16 v8bf;
typedef __attribute__((ext_vector_type(8)))  float  v8f;

#define EMB   1024
#define NH    16
#define HD    64
#define BB    4
#define SS    2048
#define MTOT  (BB * SS)          // 8192 tokens

// ---------------------------------------------------------------- helpers

__device__ __forceinline__ v8f zero8() {
  v8f r;
#pragma unroll
  for (int i = 0; i < 8; ++i) r[i] = 0.0f;
  return r;
}

__device__ __forceinline__ v16bf cat16(v8bf lo, v8bf hi) {
  v16bf r;
#pragma unroll
  for (int i = 0; i < 8; ++i) { r[i] = lo[i]; r[i + 8] = hi[i]; }
  return r;
}

__device__ __forceinline__ unsigned short f2bf(float f) {
  bf16_t h = (bf16_t)f;
  return __builtin_bit_cast(unsigned short, h);
}

// A fragment (16xK=32, 16-bit): lane m (0..15) holds row m, K {0..7,16..23};
// lane 16+m holds K {8..15,24..31}.  Source row-major, K contiguous.
__device__ __forceinline__ v16bf load_a(const bf16_t* base, int stride) {
  int lane = threadIdx.x & 31;
  int m  = lane & 15;
  int ks = (lane >> 4) << 3;              // 0 or 8
  const bf16_t* p = base + (size_t)m * stride + ks;
  v8bf lo = *reinterpret_cast<const v8bf*>(p);
  v8bf hi = *reinterpret_cast<const v8bf*>(p + 16);
  return cat16(lo, hi);
}

// B fragment (K=32 x 16, 16-bit): lane n (0..15) holds column n, K {0..7,16..23};
// lane 16+n holds K {8..15,24..31}.  Source: row n of memory == column n of B,
// K contiguous (stride = row pitch).
__device__ __forceinline__ v16bf load_b(const bf16_t* base, int stride) {
  int lane = threadIdx.x & 31;
  int n  = lane & 15;
  int ks = (lane >> 4) << 3;              // 0 or 8
  const bf16_t* p = base + (size_t)n * stride + ks;
  v8bf lo = *reinterpret_cast<const v8bf*>(p);
  v8bf hi = *reinterpret_cast<const v8bf*>(p + 16);
  return cat16(lo, hi);
}

__device__ __forceinline__ v8f wmma_bf16(v16bf a, v16bf b, v8f c) {
  return __builtin_amdgcn_wmma_f32_16x16x32_bf16(
      /*neg_a=*/false, a, /*neg_b=*/false, b,
      /*c_mod=*/(short)0, c, /*reuse_a=*/false, /*reuse_b=*/false);
}

// ---------------------------------------------------------------- stage 1

__global__ void cvt_f32_bf16(const float* __restrict__ src,
                             unsigned short* __restrict__ dst, int n) {
  int i = blockIdx.x * blockDim.x + threadIdx.x;
  if (i < n) dst[i] = f2bf(src[i]);
}

// ---------------------------------------------------------------- stage 2
// QKV projection: q[t,o] = sum_e x[t,e] * w[o,e].  One wave -> 64x64 tile.
// blockIdx.z selects Q/K/V.  Q,K stored [B,H,S,D]; V stored transposed [B,H,D,S].

__global__ void __launch_bounds__(128) qkv_proj(
    const unsigned short* __restrict__ xb,
    const unsigned short* __restrict__ wqb,
    const unsigned short* __restrict__ wkb,
    const unsigned short* __restrict__ wvb,
    unsigned short* __restrict__ qo,
    unsigned short* __restrict__ ko,
    unsigned short* __restrict__ vto) {
  const bf16_t* X = (const bf16_t*)xb;
  const unsigned short* wsel = (blockIdx.z == 0) ? wqb
                             : (blockIdx.z == 1) ? wkb : wvb;
  const bf16_t* W = (const bf16_t*)wsel;

  int wave = threadIdx.x >> 5;
  int tile = blockIdx.x * 4 + wave;       // 0..2047
  int tm = tile >> 4;                     // 0..127  (M tiles of 64)
  int tn = tile & 15;                     // 0..15   (N tiles of 64)
  int m0 = tm * 64, n0 = tn * 64;

  v8f acc[4][4];
#pragma unroll
  for (int i = 0; i < 4; ++i)
#pragma unroll
    for (int j = 0; j < 4; ++j) acc[i][j] = zero8();

  for (int k0 = 0; k0 < EMB; k0 += 32) {
    v16bf a[4], bfr[4];
#pragma unroll
    for (int i = 0; i < 4; ++i)
      a[i] = load_a(X + (size_t)(m0 + i * 16) * EMB + k0, EMB);
#pragma unroll
    for (int j = 0; j < 4; ++j)
      bfr[j] = load_b(W + (size_t)(n0 + j * 16) * EMB + k0, EMB);
#pragma unroll
    for (int i = 0; i < 4; ++i)
#pragma unroll
      for (int j = 0; j < 4; ++j)
        acc[i][j] = wmma_bf16(a[i], bfr[j], acc[i][j]);
  }

  int lane = threadIdx.x & 31;
  int nn = lane & 15, half = lane >> 4;
  int h = n0 >> 6;                        // tile spans exactly one head

#pragma unroll
  for (int i = 0; i < 4; ++i) {
#pragma unroll
    for (int r = 0; r < 8; ++r) {
      int t = m0 + i * 16 + r + half * 8; // global token row
      int bidx = t >> 11;                 // / SS
      int s = t & (SS - 1);
#pragma unroll
      for (int j = 0; j < 4; ++j) {
        int d = j * 16 + nn;
        unsigned short val = f2bf(acc[i][j][r]);
        if (blockIdx.z == 2) {
          vto[(((size_t)bidx * NH + h) * HD + d) * SS + s] = val;   // V^T
        } else {
          unsigned short* o = (blockIdx.z == 0) ? qo : ko;
          o[(((size_t)bidx * NH + h) * SS + s) * HD + d] = val;
        }
      }
    }
  }
}

// ---------------------------------------------------------------- stage 3
// Flash attention: one wave handles 16 query rows of one (b,h); online softmax
// over causal KV tiles of 64.  P re-shaped through per-wave LDS for the PV WMMA.

__global__ void __launch_bounds__(128) flash_attn(
    const unsigned short* __restrict__ qb,
    const unsigned short* __restrict__ kb,
    const unsigned short* __restrict__ vtb,
    const int* __restrict__ amask,
    unsigned short* __restrict__ ob) {
  __shared__ __align__(16) bf16_t sP[4 * 16 * 64];   // 2KB per wave

  int wave = threadIdx.x >> 5;
  int wid = blockIdx.x * 4 + wave;        // 0..8191
  int qt = wid & (SS / 16 - 1);           // 128 q-tiles per (b,h)
  int bh = wid >> 7;
  int h = bh & (NH - 1);
  int b = bh >> 4;
  int m0 = qt * 16;

  const bf16_t* Q  = (const bf16_t*)qb  + (size_t)bh * SS * HD;
  const bf16_t* K  = (const bf16_t*)kb  + (size_t)bh * SS * HD;
  const bf16_t* Vt = (const bf16_t*)vtb + (size_t)bh * HD * SS;
  bf16_t* pl = sP + wave * 16 * 64;

  int lane = threadIdx.x & 31;
  int nn = lane & 15, half = lane >> 4;

  // Q fragments for the whole sweep (D=64 -> 2 K-steps of 32)
  v16bf aq0 = load_a(Q + (size_t)m0 * HD + 0,  HD);
  v16bf aq1 = load_a(Q + (size_t)m0 * HD + 32, HD);

  float rmax[8], rsum[8];
  v8f ao[4];
#pragma unroll
  for (int r = 0; r < 8; ++r) { rmax[r] = -__builtin_inff(); rsum[r] = 0.0f; }
#pragma unroll
  for (int j = 0; j < 4; ++j) ao[j] = zero8();

  int ktlast = (m0 + 15) >> 6;            // causal tile limit
  for (int kt = 0; kt <= ktlast; ++kt) {
    int kv0 = kt * 64;

    // ---- S = Q K^T (scaled later)
    v8f sc[4];
#pragma unroll
    for (int j = 0; j < 4; ++j) {
      v16bf bk0 = load_b(K + (size_t)(kv0 + j * 16) * HD + 0,  HD);
      v16bf bk1 = load_b(K + (size_t)(kv0 + j * 16) * HD + 32, HD);
      v8f c = zero8();
      c = wmma_bf16(aq0, bk0, c);
      c = wmma_bf16(aq1, bk1, c);
      sc[j] = c;
    }

    // ---- scale + causal & padding mask
#pragma unroll
    for (int j = 0; j < 4; ++j) {
      int col = kv0 + j * 16 + nn;
      bool pad = (amask[b * SS + col] == 0);
#pragma unroll
      for (int r = 0; r < 8; ++r) {
        int row = m0 + r + half * 8;
        float v = sc[j][r] * 0.125f;      // 1/sqrt(64)
        sc[j][r] = (pad || (col > row)) ? -__builtin_inff() : v;
      }
    }

    // ---- online softmax (row stats across 16-lane halves)
#pragma unroll
    for (int r = 0; r < 8; ++r) {
      float m = sc[0][r];
#pragma unroll
      for (int j = 1; j < 4; ++j) m = fmaxf(m, sc[j][r]);
      m = fmaxf(m, __shfl_xor(m, 1, 32));
      m = fmaxf(m, __shfl_xor(m, 2, 32));
      m = fmaxf(m, __shfl_xor(m, 4, 32));
      m = fmaxf(m, __shfl_xor(m, 8, 32));
      float mnew  = fmaxf(rmax[r], m);
      float alpha = __expf(rmax[r] - mnew);
      rmax[r] = mnew;
      float s = 0.0f;
#pragma unroll
      for (int j = 0; j < 4; ++j) {
        float p = __expf(sc[j][r] - mnew);
        sc[j][r] = p;
        s += p;
      }
      s += __shfl_xor(s, 1, 32);
      s += __shfl_xor(s, 2, 32);
      s += __shfl_xor(s, 4, 32);
      s += __shfl_xor(s, 8, 32);
      rsum[r] = rsum[r] * alpha + s;
#pragma unroll
      for (int j = 0; j < 4; ++j) ao[j][r] *= alpha;
    }

    // ---- stage P into LDS (bf16, row-major 16x64), reload in A layout
#pragma unroll
    for (int j = 0; j < 4; ++j)
#pragma unroll
      for (int r = 0; r < 8; ++r)
        pl[(r + half * 8) * 64 + j * 16 + nn] = (bf16_t)sc[j][r];

    asm volatile("s_wait_dscnt 0" ::: "memory");   // wave-local DS ordering

    v16bf ap0 = load_a(pl + 0,  64);
    v16bf ap1 = load_a(pl + 32, 64);

    // ---- O += P V  (V^T layout -> contiguous B fragments)
#pragma unroll
    for (int j = 0; j < 4; ++j) {
      v16bf bv0 = load_b(Vt + (size_t)(j * 16) * SS + kv0,      SS);
      v16bf bv1 = load_b(Vt + (size_t)(j * 16) * SS + kv0 + 32, SS);
      ao[j] = wmma_bf16(ap0, bv0, ao[j]);
      ao[j] = wmma_bf16(ap1, bv1, ao[j]);
    }
  }

  // ---- normalize + store merged heads [B,S,E] as bf16 for the O-projection
#pragma unroll
  for (int r = 0; r < 8; ++r) {
    float inv = 1.0f / rsum[r];
    int s = m0 + r + half * 8;
#pragma unroll
    for (int j = 0; j < 4; ++j) {
      int d = j * 16 + nn;
      ob[((size_t)b * SS + s) * EMB + h * HD + d] = f2bf(ao[j][r] * inv);
    }
  }
}

// ---------------------------------------------------------------- stage 4
// out = attn_out @ w_o^T, f32 output.  One wave -> 64x64 tile.

__global__ void __launch_bounds__(128) out_proj(
    const unsigned short* __restrict__ ob,
    const unsigned short* __restrict__ wob,
    float* __restrict__ out) {
  const bf16_t* X = (const bf16_t*)ob;
  const bf16_t* W = (const bf16_t*)wob;

  int wave = threadIdx.x >> 5;
  int tile = blockIdx.x * 4 + wave;
  int tm = tile >> 4, tn = tile & 15;
  int m0 = tm * 64, n0 = tn * 64;

  v8f acc[4][4];
#pragma unroll
  for (int i = 0; i < 4; ++i)
#pragma unroll
    for (int j = 0; j < 4; ++j) acc[i][j] = zero8();

  for (int k0 = 0; k0 < EMB; k0 += 32) {
    v16bf a[4], bfr[4];
#pragma unroll
    for (int i = 0; i < 4; ++i)
      a[i] = load_a(X + (size_t)(m0 + i * 16) * EMB + k0, EMB);
#pragma unroll
    for (int j = 0; j < 4; ++j)
      bfr[j] = load_b(W + (size_t)(n0 + j * 16) * EMB + k0, EMB);
#pragma unroll
    for (int i = 0; i < 4; ++i)
#pragma unroll
      for (int j = 0; j < 4; ++j)
        acc[i][j] = wmma_bf16(a[i], bfr[j], acc[i][j]);
  }

  int lane = threadIdx.x & 31;
  int nn = lane & 15, half = lane >> 4;
#pragma unroll
  for (int i = 0; i < 4; ++i)
#pragma unroll
    for (int r = 0; r < 8; ++r) {
      int t = m0 + i * 16 + r + half * 8;
#pragma unroll
      for (int j = 0; j < 4; ++j)
        out[(size_t)t * EMB + n0 + j * 16 + nn] = acc[i][j][r];
    }
}

// ---------------------------------------------------------------- launch

extern "C" void kernel_launch(void* const* d_in, const int* in_sizes, int n_in,
                              void* d_out, int out_size, void* d_ws, size_t ws_size,
                              hipStream_t stream) {
  const float* x  = (const float*)d_in[0];
  const int*   am = (const int*)d_in[1];
  const float* wq = (const float*)d_in[2];
  const float* wk = (const float*)d_in[3];
  const float* wv = (const float*)d_in[4];
  const float* wo = (const float*)d_in[5];

  const size_t nx = (size_t)MTOT * EMB;   // 8,388,608 elements
  const size_t nw = (size_t)EMB * EMB;    // 1,048,576 elements

  char* p = (char*)d_ws;
  unsigned short* xb  = (unsigned short*)p; p += nx * 2;
  unsigned short* wqb = (unsigned short*)p; p += nw * 2;
  unsigned short* wkb = (unsigned short*)p; p += nw * 2;
  unsigned short* wvb = (unsigned short*)p; p += nw * 2;
  unsigned short* wob = (unsigned short*)p; p += nw * 2;
  unsigned short* qb  = (unsigned short*)p; p += nx * 2;  // [B,H,S,D]
  unsigned short* kb  = (unsigned short*)p; p += nx * 2;  // [B,H,S,D]
  unsigned short* vtb = (unsigned short*)p; p += nx * 2;  // [B,H,D,S]
  unsigned short* obf = (unsigned short*)p; p += nx * 2;  // [B,S,E]

  // Stage 1: f32 -> bf16
  cvt_f32_bf16<<<(int)((nx + 255) / 256), 256, 0, stream>>>(x,  xb,  (int)nx);
  cvt_f32_bf16<<<(int)((nw + 255) / 256), 256, 0, stream>>>(wq, wqb, (int)nw);
  cvt_f32_bf16<<<(int)((nw + 255) / 256), 256, 0, stream>>>(wk, wkb, (int)nw);
  cvt_f32_bf16<<<(int)((nw + 255) / 256), 256, 0, stream>>>(wv, wvb, (int)nw);
  cvt_f32_bf16<<<(int)((nw + 255) / 256), 256, 0, stream>>>(wo, wob, (int)nw);

  // Stage 2: QKV projections (2048 tiles / 4 waves per block; z = Q/K/V)
  qkv_proj<<<dim3(512, 1, 3), 128, 0, stream>>>(xb, wqb, wkb, wvb, qb, kb, vtb);

  // Stage 3: flash attention (8192 waves / 4 per block)
  flash_attn<<<2048, 128, 0, stream>>>(qb, kb, vtb, am, obf);

  // Stage 4: output projection
  out_proj<<<512, 128, 0, stream>>>(obf, wob, (float*)d_out);
}
